// KnowledgeEncoding_51153060496172
// MI455X (gfx1250) — matrix operand order repaired and checked by reading, not compiled, not run on hardware
//
#include <hip/hip_runtime.h>

// ---------------- constants (B,R,N,T,H,F) = (4,10,36,10,512,2048) -------------
#define BB 4
#define RR 10
#define NN 36
#define TT 10
#define HH 512
#define FF 2048

typedef __attribute__((ext_vector_type(16))) _Float16 v16h;
typedef __attribute__((ext_vector_type(8)))  float    v8f;
typedef __attribute__((ext_vector_type(4)))  unsigned int u32x4;

union Frag { v16h v; u32x4 q[2]; };

// A fragment: 16x32 f16, row M = lane%16, elems 0..7 -> K = k0+8*kh+(0..7),
// elems 8..15 -> K = k0+8*kh+16+(0..7)   (per CDNA5 ISA 7.12.2 table)
__device__ __forceinline__ v16h load_a_frag(const _Float16* buf, int ldm,
                                            int m0, int k0, int lrow, int kh) {
  Frag u;
  const _Float16* p = buf + (size_t)(m0 + lrow) * ldm + k0 + 8 * kh;
  u.q[0] = *(const u32x4*)(p);
  u.q[1] = *(const u32x4*)(p + 16);
  return u.v;
}

// B fragment: 32x16 f16, col N = lane%16, elems 0..7 -> K = k0+8*kh+(0..7), elems 8..15 -> K = k0+8*kh+16+(0..7)
// (weight stored row-major W[n][k], so two contiguous 16B runs per lane)
__device__ __forceinline__ v16h load_b_frag(const _Float16* w, int ldk,
                                            int n0, int k0, int lrow, int kh) {
  Frag u;
  const _Float16* p = w + (size_t)(n0 + lrow) * ldk + k0 + 8 * kh;
  u.q[0] = *(const u32x4*)(p);
  u.q[1] = *(const u32x4*)(p + 16);
  return u.v;
}

__device__ __forceinline__ v8f wmma16(v16h a, v16h b, v8f c) {
  return __builtin_amdgcn_wmma_f32_16x16x32_f16(false, a, false, b,
                                                (short)0, c, false, false);
}

// ------------------------- prep kernels --------------------------------------
__global__ void k_conv_f16(const float* __restrict__ src,
                           _Float16* __restrict__ dst, int n) {
  int i = blockIdx.x * 256 + threadIdx.x;
  if (i < n) dst[i] = (_Float16)src[i];
}

// qv/qt: (B*R,H) = ques @ w.T + b
__global__ void k_q(const float* __restrict__ ques,
                    const float* __restrict__ wv1_w, const float* __restrict__ wv1_b,
                    const float* __restrict__ wt1_w, const float* __restrict__ wt1_b,
                    float* __restrict__ qv, float* __restrict__ qt) {
  int idx = blockIdx.x * 256 + threadIdx.x;
  const int TOT = BB * RR * HH;
  if (idx >= 2 * TOT) return;
  int which = idx / TOT;
  int rem   = idx % TOT;
  int br = rem / HH, h = rem % HH;
  const float* w = (which ? wt1_w : wv1_w) + (size_t)h * HH;
  const float* x = ques + (size_t)br * HH;
  float acc = (which ? wt1_b : wv1_b)[h];
  for (int k = 0; k < HH; ++k) acc += x[k] * w[k];
  (which ? qt : qv)[rem] = acc;
}

// Factored gating vectors:  u = w_gate ⊙ q ;  a = u@W_left ; c = u@W_right ; d = u·b3 + b_gate
__global__ void k_acd(const float* __restrict__ qv, const float* __restrict__ qt,
                      const float* __restrict__ wv_w, const float* __restrict__ wv_b,
                      const float* __restrict__ wv3_w, const float* __restrict__ wv3_b,
                      const float* __restrict__ wt_w, const float* __restrict__ wt_b,
                      const float* __restrict__ wt3_w, const float* __restrict__ wt3_b,
                      float* a_v, float* c_v, float* d_v,
                      float* a_t, float* c_t, float* d_t) {
  int idx = blockIdx.x * 256 + threadIdx.x;
  const int LDV = FF + HH;   // 2560
  const int LDT = 2 * HH;    // 1024
  if (idx < BB * RR * FF) {                     // a_v
    int br = idx >> 11, f = idx & (FF - 1);
    const float* q = qv + (size_t)br * HH; float acc = 0.f;
    for (int h = 0; h < HH; ++h) acc += wv_w[h] * q[h] * wv3_w[(size_t)h * LDV + f];
    a_v[idx] = acc; return;
  }
  idx -= BB * RR * FF;
  if (idx < BB * RR * HH) {                     // c_v
    int br = idx >> 9, k = idx & (HH - 1);
    const float* q = qv + (size_t)br * HH; float acc = 0.f;
    for (int h = 0; h < HH; ++h) acc += wv_w[h] * q[h] * wv3_w[(size_t)h * LDV + FF + k];
    c_v[idx] = acc; return;
  }
  idx -= BB * RR * HH;
  if (idx < BB * RR * HH) {                     // a_t
    int br = idx >> 9, j = idx & (HH - 1);
    const float* q = qt + (size_t)br * HH; float acc = 0.f;
    for (int h = 0; h < HH; ++h) acc += wt_w[h] * q[h] * wt3_w[(size_t)h * LDT + j];
    a_t[idx] = acc; return;
  }
  idx -= BB * RR * HH;
  if (idx < BB * RR * HH) {                     // c_t
    int br = idx >> 9, k = idx & (HH - 1);
    const float* q = qt + (size_t)br * HH; float acc = 0.f;
    for (int h = 0; h < HH; ++h) acc += wt_w[h] * q[h] * wt3_w[(size_t)h * LDT + HH + k];
    c_t[idx] = acc; return;
  }
  idx -= BB * RR * HH;
  if (idx < BB * RR) {                          // d_v
    const float* q = qv + (size_t)idx * HH; float acc = wv_b[0];
    for (int h = 0; h < HH; ++h) acc += wv_w[h] * q[h] * wv3_b[h];
    d_v[idx] = acc; return;
  }
  idx -= BB * RR;
  if (idx < BB * RR) {                          // d_t
    const float* q = qt + (size_t)idx * HH; float acc = wt_b[0];
    for (int h = 0; h < HH; ++h) acc += wt_w[h] * q[h] * wt3_b[h];
    d_t[idx] = acc; return;
  }
}

// ------------------------- vr precompute (r-independent) ---------------------
// One WG per (b,n1): vr[36x512] = Vrel @ wv2^T + b -> f16 workspace.
// B-fragment reused across 3 M-tiles; kk unroll bounded to avoid VGPR spill.
__global__ void k_vr(const float* __restrict__ vrel,
                     const _Float16* __restrict__ wv2bf,
                     const float* __restrict__ wv2_b,
                     _Float16* __restrict__ vrbf) {
  extern __shared__ char smem[];
  _Float16* abuf = (_Float16*)smem;           // 48x512 f16
  const int wg = blockIdx.x;                  // b*NN + n1
  const int tid = threadIdx.x;
  const int lane = tid & 31, wid = tid >> 5;
  const int lrow = lane & 15, kh = lane >> 4;

  const float* Asl = vrel + (size_t)wg * NN * HH;
  for (int i = tid; i < 48 * HH; i += 256) {
    int m = i >> 9, k = i & (HH - 1);
    abuf[i] = (_Float16)((m < NN) ? Asl[(size_t)m * HH + k] : 0.f);
  }
  __syncthreads();

  _Float16* op = vrbf + (size_t)wg * NN * HH;
  for (int nt = wid; nt < 32; nt += 8) {
    int n0 = nt * 16;
    v8f acc0 = {}, acc1 = {}, acc2 = {};
#pragma unroll 2
    for (int kk = 0; kk < HH; kk += 32) {
      v16h w  = load_b_frag(wv2bf, HH, n0, kk, lrow, kh);
      v16h a0 = load_a_frag(abuf, HH, 0,  kk, lrow, kh);
      v16h a1 = load_a_frag(abuf, HH, 16, kk, lrow, kh);
      v16h a2 = load_a_frag(abuf, HH, 32, kk, lrow, kh);
      acc0 = wmma16(a0, w, acc0);
      acc1 = wmma16(a1, w, acc1);
      acc2 = wmma16(a2, w, acc2);
    }
    int h = n0 + lrow;
    float bias = wv2_b[h];
#pragma unroll
    for (int i = 0; i < 8; ++i) {
      int m = i + 8 * kh;
      op[(size_t)m * HH + h]        = (_Float16)(acc0[i] + bias);
      op[(size_t)(m + 16) * HH + h] = (_Float16)(acc1[i] + bias);
      if (m + 32 < NN)
        op[(size_t)(m + 32) * HH + h] = (_Float16)(acc2[i] + bias);
    }
  }
}

// ------------------------- visual branch -------------------------------------
// One WG per (b,r,n1): stage vr (f16), proj = qv⊙vr, logits = proj @ wve^T,
// column softmax over n2, gating, out = img * Σsoftmax.
#define V_LBUF   0                      // 48x512 f16 (logits)
#define V_VRBUF  49152                  // 48x512 f16
#define V_PRBUF  98304                  // 48x512 f16
#define V_QVS    147456                 // 512 f32
#define V_CVS    149504                 // 512 f32
#define V_GBUF   151552                 // 48 f32
#define V_RED    151744                 // 256 f32
#define V_SV     152768                 // 4 f32
#define V_SMEM   152832

__global__ void k_visual(const float* __restrict__ img,
                         const _Float16* __restrict__ vrbf,
                         const float* __restrict__ qv,
                         const float* __restrict__ a_v,
                         const float* __restrict__ c_v,
                         const float* __restrict__ d_v,
                         const _Float16* __restrict__ wvebf,
                         const float* __restrict__ wve_b,
                         float* __restrict__ out_v) {
  extern __shared__ char smem[];
  _Float16* lbuf  = (_Float16*)(smem + V_LBUF);
  _Float16* vrbuf = (_Float16*)(smem + V_VRBUF);
  _Float16* prbuf = (_Float16*)(smem + V_PRBUF);
  float* qv_s = (float*)(smem + V_QVS);
  float* cv_s = (float*)(smem + V_CVS);
  float* gbuf = (float*)(smem + V_GBUF);
  float* red  = (float*)(smem + V_RED);
  float* sv   = (float*)(smem + V_SV);

  const int wg = blockIdx.x;               // (b*RR+r)*NN + n1
  const int n1 = wg % NN;
  const int br = wg / NN;
  const int b  = br / RR;
  const int tid = threadIdx.x;
  const int lane = tid & 31, wid = tid >> 5;
  const int lrow = lane & 15, kh = lane >> 4;

  const float* imgp = img + (size_t)(b * NN + n1) * FF;

  for (int h = tid; h < HH; h += 256) {
    qv_s[h] = qv[(size_t)br * HH + h];
    cv_s[h] = c_v[(size_t)br * HH + h];
  }
  if (tid < 48) gbuf[tid] = 0.f;
  __syncthreads();

  // stage vr slice (f16, L2-resident) + proj = qv*vr, zero-pad M to 48
  const _Float16* vsl = vrbf + (size_t)(b * NN + n1) * NN * HH;
  for (int i = tid; i < 48 * HH; i += 256) {
    int m = i >> 9, k = i & (HH - 1);
    float v = (m < NN) ? (float)vsl[(size_t)m * HH + k] : 0.f;
    vrbuf[i] = (_Float16)v;
    prbuf[i] = (_Float16)(qv_s[k] * v);
  }
  __syncthreads();

  // GEMM: logits = proj @ wve^T + b  (B fragment reused across 3 M tiles)
  for (int nt = wid; nt < 32; nt += 8) {
    int n0 = nt * 16;
    v8f acc0 = {}, acc1 = {}, acc2 = {};
#pragma unroll 2
    for (int kk = 0; kk < HH; kk += 32) {
      v16h w  = load_b_frag(wvebf, HH, n0, kk, lrow, kh);
      v16h a0 = load_a_frag(prbuf, HH, 0,  kk, lrow, kh);
      v16h a1 = load_a_frag(prbuf, HH, 16, kk, lrow, kh);
      v16h a2 = load_a_frag(prbuf, HH, 32, kk, lrow, kh);
      acc0 = wmma16(a0, w, acc0);
      acc1 = wmma16(a1, w, acc1);
      acc2 = wmma16(a2, w, acc2);
    }
    int h = n0 + lrow;
    float bias = wve_b[h];
#pragma unroll
    for (int i = 0; i < 8; ++i) {
      int m = i + 8 * kh;
      lbuf[(m)      * HH + h] = (_Float16)(acc0[i] + bias);
      lbuf[(m + 16) * HH + h] = (_Float16)(acc1[i] + bias);
      lbuf[(m + 32) * HH + h] = (_Float16)(acc2[i] + bias);
    }
  }
  __syncthreads();

  // column softmax over n2 (M dim), vrw = relw*vr, g[m] += c_v[h]*vrw
  for (int h = tid; h < HH; h += 256) {
    float mx = -1e30f;
    for (int m = 0; m < NN; ++m) mx = fmaxf(mx, (float)lbuf[m * HH + h]);
    float s = 0.f;
    for (int m = 0; m < NN; ++m) s += __expf((float)lbuf[m * HH + h] - mx);
    float inv = 1.f / s, ch = cv_s[h];
    for (int m = 0; m < NN; ++m) {
      float w = __expf((float)lbuf[m * HH + h] - mx) * inv;
      atomicAdd(&gbuf[m], ch * w * (float)vrbuf[m * HH + h]);
    }
  }
  __syncthreads();

  // pdot = a_v[br] . img[b,n1]
  float p = 0.f;
  const float* avp = a_v + (size_t)br * FF;
  for (int f = tid; f < FF; f += 256) p += avp[f] * imgp[f];
  red[tid] = p; __syncthreads();
  for (int s = 128; s > 0; s >>= 1) {
    if (tid < s) red[tid] += red[tid + s];
    __syncthreads();
  }
  if (tid == 0) {
    float pd = red[0] + d_v[br];
    float mx = -1e30f;
    for (int m = 0; m < NN; ++m) { float x = gbuf[m] + pd; gbuf[m] = x; mx = fmaxf(mx, x); }
    float ssum = 0.f;
    for (int m = 0; m < NN; ++m) ssum += __expf(gbuf[m] - mx);
    float inv = 1.f / ssum, S = 0.f;
    for (int m = 0; m < NN; ++m) S += __expf(gbuf[m] - mx) * inv;  // Σ softmax (≈1)
    sv[0] = S;
  }
  __syncthreads();
  float S = sv[0];
  float* op = out_v + (size_t)wg * FF;
  for (int f = tid; f < FF; f += 256) op[f] = S * imgp[f];
}

// ------------------------- text branch ---------------------------------------
#define T_ABUF   0                      // 16x1024 f16 (A stage, reused for logits 16x512)
#define T_TRBUF  32768                  // 16x512 f16
#define T_PRBUF  49152                  // 16x512 f16
#define T_QTS    65536                  // 512 f32
#define T_CTS    67584                  // 512 f32
#define T_GBUF   69632                  // 16 f32
#define T_RED    69760                  // 256 f32
#define T_SV     70784                  // 4 f32
#define T_SMEM   70912

__global__ void k_text(const float* __restrict__ hist,
                       const float* __restrict__ trel,
                       const float* __restrict__ qt,
                       const float* __restrict__ a_t,
                       const float* __restrict__ c_t,
                       const float* __restrict__ d_t,
                       const _Float16* __restrict__ wt2bf,
                       const float* __restrict__ wt2_b,
                       const _Float16* __restrict__ wtebf,
                       const float* __restrict__ wte_b,
                       float* __restrict__ out_t) {
  extern __shared__ char smem[];
  _Float16* abuf  = (_Float16*)(smem + T_ABUF);
  _Float16* trbuf = (_Float16*)(smem + T_TRBUF);
  _Float16* prbuf = (_Float16*)(smem + T_PRBUF);
  float* qt_s = (float*)(smem + T_QTS);
  float* ct_s = (float*)(smem + T_CTS);
  float* gbuf = (float*)(smem + T_GBUF);
  float* red  = (float*)(smem + T_RED);
  float* sv   = (float*)(smem + T_SV);

  const int K1 = 2 * HH;                   // 1024
  const int wg = blockIdx.x;               // [0, B*R*T)
  const int br = wg / TT;
  const int tid = threadIdx.x;
  const int lane = tid & 31, wid = tid >> 5;
  const int lrow = lane & 15, kh = lane >> 4;

  const float* Asl   = trel + (size_t)wg * RR * K1;   // [10][1024]
  const float* histp = hist + (size_t)wg * HH;

  for (int i = tid; i < 16 * K1; i += 256) {
    int m = i >> 10, k = i & (K1 - 1);
    abuf[i] = (_Float16)((m < RR) ? Asl[(size_t)m * K1 + k] : 0.f);
  }
  for (int h = tid; h < HH; h += 256) {
    qt_s[h] = qt[(size_t)br * HH + h];
    ct_s[h] = c_t[(size_t)br * HH + h];
  }
  if (tid < 16) gbuf[tid] = 0.f;
  __syncthreads();

  // GEMM1: tr = A[16x1024] @ wt2^T + b  (32 N-tiles, K=1024)
  for (int t = wid; t < 32; t += 8) {
    int n0 = t * 16;
    v8f acc = {};
#pragma unroll 2
    for (int kk = 0; kk < K1; kk += 32) {
      v16h a = load_a_frag(abuf, K1, 0, kk, lrow, kh);
      v16h w = load_b_frag(wt2bf, K1, n0, kk, lrow, kh);
      acc = wmma16(a, w, acc);
    }
    int h = n0 + lrow;
    float bias = wt2_b[h], q = qt_s[h];
#pragma unroll
    for (int i = 0; i < 8; ++i) {
      int m = i + 8 * kh;
      float trv = acc[i] + bias;
      trbuf[m * HH + h] = (_Float16)trv;
      prbuf[m * HH + h] = (_Float16)(q * trv);
    }
  }
  __syncthreads();

  // GEMM2: logits = proj @ wte^T + b -> abuf (16x512, reuse)
  for (int t = wid; t < 32; t += 8) {
    int n0 = t * 16;
    v8f acc = {};
#pragma unroll 2
    for (int kk = 0; kk < HH; kk += 32) {
      v16h a = load_a_frag(prbuf, HH, 0, kk, lrow, kh);
      v16h w = load_b_frag(wtebf, HH, n0, kk, lrow, kh);
      acc = wmma16(a, w, acc);
    }
    int h = n0 + lrow;
    float bias = wte_b[h];
#pragma unroll
    for (int i = 0; i < 8; ++i) {
      int m = i + 8 * kh;
      abuf[m * HH + h] = (_Float16)(acc[i] + bias);
    }
  }
  __syncthreads();

  for (int h = tid; h < HH; h += 256) {
    float mx = -1e30f;
    for (int m = 0; m < RR; ++m) mx = fmaxf(mx, (float)abuf[m * HH + h]);
    float s = 0.f;
    for (int m = 0; m < RR; ++m) s += __expf((float)abuf[m * HH + h] - mx);
    float inv = 1.f / s, ch = ct_s[h];
    for (int m = 0; m < RR; ++m) {
      float w = __expf((float)abuf[m * HH + h] - mx) * inv;
      atomicAdd(&gbuf[m], ch * w * (float)trbuf[m * HH + h]);
    }
  }
  __syncthreads();

  float p = 0.f;
  const float* atp = a_t + (size_t)br * HH;
  for (int h = tid; h < HH; h += 256) p += atp[h] * histp[h];
  red[tid] = p; __syncthreads();
  for (int s = 128; s > 0; s >>= 1) {
    if (tid < s) red[tid] += red[tid + s];
    __syncthreads();
  }
  if (tid == 0) {
    float pd = red[0] + d_t[br];
    float mx = -1e30f;
    for (int m = 0; m < RR; ++m) { float x = gbuf[m] + pd; gbuf[m] = x; mx = fmaxf(mx, x); }
    float ssum = 0.f;
    for (int m = 0; m < RR; ++m) ssum += __expf(gbuf[m] - mx);
    float inv = 1.f / ssum, S = 0.f;
    for (int m = 0; m < RR; ++m) S += __expf(gbuf[m] - mx) * inv;
    sv[0] = S;
  }
  __syncthreads();
  float S = sv[0];
  float* op = out_t + (size_t)wg * HH;
  for (int h = tid; h < HH; h += 256) op[h] = S * histp[h];
}

// ------------------------- launch --------------------------------------------
extern "C" void kernel_launch(void* const* d_in, const int* in_sizes, int n_in,
                              void* d_out, int out_size, void* d_ws, size_t ws_size,
                              hipStream_t stream) {
  (void)in_sizes; (void)n_in; (void)out_size; (void)ws_size;
  const float* img   = (const float*)d_in[0];
  const float* ques  = (const float*)d_in[1];
  const float* vrel  = (const float*)d_in[2];
  const float* hist  = (const float*)d_in[3];
  const float* trel  = (const float*)d_in[4];
  const float* wv1_w = (const float*)d_in[5];
  const float* wv1_b = (const float*)d_in[6];
  const float* wv2_w = (const float*)d_in[7];
  const float* wv2_b = (const float*)d_in[8];
  const float* wve_w = (const float*)d_in[9];
  const float* wve_b = (const float*)d_in[10];
  const float* wt1_w = (const float*)d_in[11];
  const float* wt1_b = (const float*)d_in[12];
  const float* wt2_w = (const float*)d_in[13];
  const float* wt2_b = (const float*)d_in[14];
  const float* wte_w = (const float*)d_in[15];
  const float* wte_b = (const float*)d_in[16];
  const float* wv3_w = (const float*)d_in[17];
  const float* wv3_b = (const float*)d_in[18];
  const float* wv_w  = (const float*)d_in[19];
  const float* wv_b  = (const float*)d_in[20];
  const float* wt3_w = (const float*)d_in[21];
  const float* wt3_b = (const float*)d_in[22];
  const float* wt_w  = (const float*)d_in[23];
  const float* wt_b  = (const float*)d_in[24];

  float* out_v = (float*)d_out;
  float* out_t = out_v + (size_t)BB * RR * NN * FF;

  // workspace carve
  char* w = (char*)d_ws;
  _Float16* wv2bf = (_Float16*)w; w += (size_t)HH * HH * 2;
  _Float16* wvebf = (_Float16*)w; w += (size_t)HH * HH * 2;
  _Float16* wt2bf = (_Float16*)w; w += (size_t)HH * 2 * HH * 2;
  _Float16* wtebf = (_Float16*)w; w += (size_t)HH * HH * 2;
  _Float16* vrbf  = (_Float16*)w; w += (size_t)BB * NN * NN * HH * 2;  // 5.3 MB
  float* qv  = (float*)w; w += (size_t)BB * RR * HH * 4;
  float* qt  = (float*)w; w += (size_t)BB * RR * HH * 4;
  float* a_v = (float*)w; w += (size_t)BB * RR * FF * 4;
  float* c_v = (float*)w; w += (size_t)BB * RR * HH * 4;
  float* a_t = (float*)w; w += (size_t)BB * RR * HH * 4;
  float* c_t = (float*)w; w += (size_t)BB * RR * HH * 4;
  float* d_v = (float*)w; w += 256;
  float* d_t = (float*)w; w += 256;

  hipFuncSetAttribute((const void*)k_visual,
                      hipFuncAttributeMaxDynamicSharedMemorySize, V_SMEM);
  hipFuncSetAttribute((const void*)k_text,
                      hipFuncAttributeMaxDynamicSharedMemorySize, T_SMEM);

  k_conv_f16<<<(HH * HH + 255) / 256, 256, 0, stream>>>(wv2_w, wv2bf, HH * HH);
  k_conv_f16<<<(HH * HH + 255) / 256, 256, 0, stream>>>(wve_w, wvebf, HH * HH);
  k_conv_f16<<<(HH * 2 * HH + 255) / 256, 256, 0, stream>>>(wt2_w, wt2bf, HH * 2 * HH);
  k_conv_f16<<<(HH * HH + 255) / 256, 256, 0, stream>>>(wte_w, wtebf, HH * HH);

  k_q<<<(2 * BB * RR * HH + 255) / 256, 256, 0, stream>>>(
      ques, wv1_w, wv1_b, wt1_w, wt1_b, qv, qt);

  const int acd_total = BB * RR * (FF + 3 * HH + 2);
  k_acd<<<(acd_total + 255) / 256, 256, 0, stream>>>(
      qv, qt, wv_w, wv_b, wv3_w, wv3_b, wt_w, wt_b, wt3_w, wt3_b,
      a_v, c_v, d_v, a_t, c_t, d_t);

  k_vr<<<BB * NN, 256, 48 * HH * 2, stream>>>(vrel, wv2bf, wv2_b, vrbf);

  k_visual<<<BB * RR * NN, 256, V_SMEM, stream>>>(
      img, vrbf, qv, a_v, c_v, d_v, wvebf, wve_b, out_v);

  k_text<<<BB * RR * TT, 256, T_SMEM, stream>>>(
      hist, trel, qt, a_t, c_t, d_t, wt2bf, wt2_b, wtebf, wte_b, out_t);
}
